// WindowSelfAttention_63041529971176
// MI455X (gfx1250) — matrix operand order repaired and checked by reading, not compiled, not run on hardware
//
#include <hip/hip_runtime.h>
#include <hip/hip_bf16.h>

typedef __attribute__((ext_vector_type(16))) __bf16 v16bf;
typedef __attribute__((ext_vector_type(8)))  __bf16 v8bf;
typedef __attribute__((ext_vector_type(8)))  float  v8f;

#define LDX 264   // Xt / Op row stride (bf16 elems), padded vs 256
#define LDXW (LDX / 2)
#define LDQ 40    // Q / K row stride, padded vs 32
#define LDV 72    // Vt row stride, padded vs 64
#define LDSB 72   // Sb row stride, padded vs 64

struct SM {
  __bf16 Xt[64 * LDX];        // phase0/1: x window [p][c]; phase2/3: Op[p][c]
  __bf16 Q [8 * 64 * LDQ];    // [h][i][d]
  __bf16 K [8 * 64 * LDQ];    // [h][j][d]
  __bf16 Vt[8 * 32 * LDV];    // [h][d][j]
  __bf16 Sb[8 * 64 * LDSB];   // softmaxed attn [h][i][j]
};                            // ~221 KB < 320 KB WGP LDS

// A-operand (16x32, M x K): lane = (m, half); K = {8*half..+7, 16+8*half..+7}
__device__ __forceinline__ v16bf loadA(const __bf16* base, int ld, int lane) {
  const int m = lane & 15, hl = lane >> 4;
  const __bf16* r = base + m * ld + hl * 8;
  v8bf lo = *(const v8bf*)(r);
  v8bf hi = *(const v8bf*)(r + 16);
  v16bf o;
#pragma unroll
  for (int t = 0; t < 8; ++t) { o[t] = lo[t]; o[t + 8] = hi[t]; }
  return o;
}

// B-operand (32x16, K x N) with transposed [N][K] storage:
// lane = (n, half); K = {8*half..+7, 16+8*half..+7}
__device__ __forceinline__ v16bf loadB(const __bf16* base, int ld, int lane) {
  const int n = lane & 15, hl = lane >> 4;
  const __bf16* r = base + n * ld + hl * 8;
  v8bf lo = *(const v8bf*)(r);
  v8bf hi = *(const v8bf*)(r + 16);
  v16bf o;
#pragma unroll
  for (int t = 0; t < 8; ++t) { o[t] = lo[t]; o[t + 8] = hi[t]; }
  return o;
}

__device__ __forceinline__ v8f wmma_bf16(v16bf a, v16bf b, v8f c) {
  return __builtin_amdgcn_wmma_f32_16x16x32_bf16(false, a, false, b,
                                                 (short)0, c, false, false);
}

// pack two f32 -> two bf16 in one dword (lowers to v_cvt_pk_bf16_f32)
__device__ __forceinline__ unsigned int pack2bf(float a, float b) {
  union { __bf16 h[2]; unsigned int u; } t;
  t.h[0] = (__bf16)a; t.h[1] = (__bf16)b;
  return t.u;
}

__global__ void convert_weights(const float* __restrict__ wq,
                                const float* __restrict__ wo,
                                __bf16* __restrict__ wq_bf,
                                __bf16* __restrict__ wo_bf) {
  const int i = blockIdx.x * blockDim.x + threadIdx.x;
  if (i < 768 * 256) wq_bf[i] = (__bf16)wq[i];
  if (i < 256 * 256) wo_bf[i] = (__bf16)wo[i];
}

__global__ void __launch_bounds__(256)
win_attn_kernel(const float* __restrict__ x,
                const __bf16* __restrict__ wq,
                const __bf16* __restrict__ wo,
                const float* __restrict__ bout,
                float* __restrict__ out) {
  __shared__ SM sm;
  const int tid  = threadIdx.x;
  // readfirstlane: provably wave-uniform -> scalar branches & SGPR addressing
  const int wave = __builtin_amdgcn_readfirstlane(tid >> 5);
  const int lane = tid & 31;
  const int hl   = lane >> 4;
  const int wi = blockIdx.x;
  const int b  = wi >> 10;
  const int wh = (wi >> 5) & 31;
  const int ww = wi & 31;

  // ---------- Phase 0: x window -> Xt[p][c] (bf16, packed b32 stores) ----------
  {
    const int y = wave;                        // 0..7 (scalar)
#pragma unroll
    for (int k = 0; k < 8; k += 2) {
      const int c = lane * 8 + k;              // even; lane varies c -> banks spread
      const float* s0 = x + (size_t)b * (256u * 256u * 256u)
                      + (size_t)c * 65536u
                      + (size_t)(wh * 8 + y) * 256u + (size_t)(ww * 8);
      const float* s1 = s0 + 65536u;           // channel c+1
      const float4 a0 = ((const float4*)s0)[0];
      const float4 a1 = ((const float4*)s0)[1];
      const float4 b0 = ((const float4*)s1)[0];
      const float4 b1 = ((const float4*)s1)[1];
      unsigned int* colp = (unsigned int*)(sm.Xt + (y * 8) * LDX + c);
      colp[0 * LDXW] = pack2bf(a0.x, b0.x);
      colp[1 * LDXW] = pack2bf(a0.y, b0.y);
      colp[2 * LDXW] = pack2bf(a0.z, b0.z);
      colp[3 * LDXW] = pack2bf(a0.w, b0.w);
      colp[4 * LDXW] = pack2bf(a1.x, b1.x);
      colp[5 * LDXW] = pack2bf(a1.y, b1.y);
      colp[6 * LDXW] = pack2bf(a1.z, b1.z);
      colp[7 * LDXW] = pack2bf(a1.w, b1.w);
    }
  }
  __syncthreads();

  // ---------- Phase 1: qkv = w_qkv(768x256) @ Xt(256x64) ----------
  for (int t6 = 0; t6 < 6; ++t6) {
    const int mt = wave * 6 + t6;               // scalar; 48 M-tiles / 8 waves
    v16bf Ak[8];
#pragma unroll
    for (int kc = 0; kc < 8; ++kc)
      Ak[kc] = loadA(wq + mt * 16 * 256 + kc * 32, 256, lane);
    for (int nt = 0; nt < 4; ++nt) {
      v8f acc = {};
#pragma unroll
      for (int kc = 0; kc < 8; ++kc) {
        v16bf Bk = loadB(&sm.Xt[nt * 16 * LDX + kc * 32], LDX, lane);
        acc = wmma_bf16(Ak[kc], Bk, acc);
      }
      const int p = nt * 16 + (lane & 15);
      // mt is scalar -> plain scalar branches, no EXEC masking.
      // A 16-row tile never crosses a 32-channel head boundary.
      if (mt < 16) {                            // ---- Q: [h][p][d], packed b32
        const int cb  = mt * 16;
        const int h   = cb >> 5;                // scalar
        const int ddb = (cb & 31) + 8 * hl;
        unsigned int* rowp =
            (unsigned int*)&sm.Q[h * 64 * LDQ + p * LDQ + ddb];
#pragma unroll
        for (int vp = 0; vp < 4; ++vp)
          rowp[vp] = pack2bf(acc[2 * vp], acc[2 * vp + 1]);
      } else if (mt < 32) {                     // ---- K: [h][p][d], packed b32
        const int cb  = (mt - 16) * 16;
        const int h   = cb >> 5;
        const int ddb = (cb & 31) + 8 * hl;
        unsigned int* rowp =
            (unsigned int*)&sm.K[h * 64 * LDQ + p * LDQ + ddb];
#pragma unroll
        for (int vp = 0; vp < 4; ++vp)
          rowp[vp] = pack2bf(acc[2 * vp], acc[2 * vp + 1]);
      } else {                                  // ---- V: [h][d][p], b16 column
        const int cb  = (mt - 32) * 16;
        const int h   = cb >> 5;
        const int ddb = (cb & 31) + 8 * hl;
        __bf16* colp = &sm.Vt[h * 32 * LDV + ddb * LDV + p];
#pragma unroll
        for (int v = 0; v < 8; ++v)
          colp[v * LDV] = (__bf16)acc[v];
      }
    }
  }
  __syncthreads();

  // ---------- Phase 2: per-head attention (wave = head, h scalar) ----------
  {
    const int h = wave;
    const __bf16* Qh = &sm.Q[h * 64 * LDQ];
    const __bf16* Kh = &sm.K[h * 64 * LDQ];
    const float scale = 0.17677669529663687f;   // 1/sqrt(32)
    const int jb = lane & 15;
    v16bf Bj[4];
#pragma unroll
    for (int jt = 0; jt < 4; ++jt) Bj[jt] = loadB(Kh + jt * 16 * LDQ, LDQ, lane);
    // Stream one 16-row block of S at a time: peak live = Bj[4] + Srow[4]
    for (int it = 0; it < 4; ++it) {
      v16bf Ai = loadA(Qh + it * 16 * LDQ, LDQ, lane);
      v8f Srow[4];
#pragma unroll
      for (int jt = 0; jt < 4; ++jt) {
        v8f z = {};
        Srow[jt] = wmma_bf16(Ai, Bj[jt], z);    // logits (unscaled)
      }
#pragma unroll
      for (int r = 0; r < 8; ++r) {
        float x0 = Srow[0][r] * scale, x1 = Srow[1][r] * scale;
        float x2 = Srow[2][r] * scale, x3 = Srow[3][r] * scale;
        float m = fmaxf(fmaxf(x0, x1), fmaxf(x2, x3));
        m = fmaxf(m, __shfl_xor(m, 1));
        m = fmaxf(m, __shfl_xor(m, 2));
        m = fmaxf(m, __shfl_xor(m, 4));
        m = fmaxf(m, __shfl_xor(m, 8));   // row lives in a 16-lane half
        float e0 = __expf(x0 - m), e1 = __expf(x1 - m);
        float e2 = __expf(x2 - m), e3 = __expf(x3 - m);
        float s = e0 + e1 + e2 + e3;
        s += __shfl_xor(s, 1);
        s += __shfl_xor(s, 2);
        s += __shfl_xor(s, 4);
        s += __shfl_xor(s, 8);
        const float inv = 1.0f / s;
        const int i = it * 16 + r + 8 * hl;
        __bf16* row = &sm.Sb[h * 64 * LDSB + i * LDSB];
        row[ 0 + jb] = (__bf16)(e0 * inv);
        row[16 + jb] = (__bf16)(e1 * inv);
        row[32 + jb] = (__bf16)(e2 * inv);
        row[48 + jb] = (__bf16)(e3 * inv);
      }
    }
    // O = softmax(S) @ V : M=64(i) x N=32(d) x K=64(j); wave-local, no barrier
    const __bf16* Vh = &sm.Vt[h * 32 * LDV];
    const __bf16* Sh = &sm.Sb[h * 64 * LDSB];
    for (int it = 0; it < 4; ++it) {
      v16bf A0 = loadA(Sh + it * 16 * LDSB +  0, LDSB, lane);
      v16bf A1 = loadA(Sh + it * 16 * LDSB + 32, LDSB, lane);
#pragma unroll
      for (int nt = 0; nt < 2; ++nt) {
        v16bf B0 = loadB(Vh + nt * 16 * LDV +  0, LDV, lane);
        v16bf B1 = loadB(Vh + nt * 16 * LDV + 32, LDV, lane);
        v8f acc = {};
        acc = wmma_bf16(A0, B0, acc);
        acc = wmma_bf16(A1, B1, acc);
        const int c = h * 32 + nt * 16 + (lane & 15);
        __bf16* colp = &sm.Xt[(it * 16 + 8 * hl) * LDX + c];  // Op[p][c]
#pragma unroll
        for (int v = 0; v < 8; ++v)
          colp[v * LDX] = (__bf16)acc[v];
      }
    }
  }
  __syncthreads();

  // ---------- Phase 3: out = w_out(256x256) @ Op(256x64) + bias ----------
  for (int t2 = 0; t2 < 2; ++t2) {
    const int mt = wave * 2 + t2;                 // scalar; 16 M-tiles / 8 waves
    v16bf Ak[8];
#pragma unroll
    for (int kc = 0; kc < 8; ++kc)
      Ak[kc] = loadA(wo + mt * 16 * 256 + kc * 32, 256, lane);
    float bias[8];
#pragma unroll
    for (int v = 0; v < 8; ++v)
      bias[v] = bout[mt * 16 + v + 8 * hl];       // once per M-tile
    for (int nt = 0; nt < 4; ++nt) {
      v8f acc = {};
#pragma unroll
      for (int kc = 0; kc < 8; ++kc) {
        v16bf Bk = loadB(&sm.Xt[nt * 16 * LDX + kc * 32], LDX, lane);
        acc = wmma_bf16(Ak[kc], Bk, acc);
      }
      const int p = nt * 16 + (lane & 15);
      const int y = p >> 3, xw = p & 7;
      float* dstp = out + (size_t)b * (256u * 256u * 256u)
                  + (size_t)(mt * 16 + 8 * hl) * 65536u
                  + (size_t)(wh * 8 + y) * 256u + (size_t)(ww * 8 + xw);
#pragma unroll
      for (int v = 0; v < 8; ++v)
        dstp[(size_t)v * 65536u] = acc[v] + bias[v];
    }
  }
}

extern "C" void kernel_launch(void* const* d_in, const int* in_sizes, int n_in,
                              void* d_out, int out_size, void* d_ws, size_t ws_size,
                              hipStream_t stream) {
  (void)in_sizes; (void)n_in; (void)out_size; (void)ws_size;
  const float* x     = (const float*)d_in[0];
  const float* w_qkv = (const float*)d_in[1];
  const float* w_out = (const float*)d_in[2];
  const float* b_out = (const float*)d_in[3];
  float* out = (float*)d_out;

  __bf16* wq_bf = (__bf16*)d_ws;                 // 768*256 bf16
  __bf16* wo_bf = wq_bf + 768 * 256;             // 256*256 bf16

  convert_weights<<<dim3((768 * 256 + 255) / 256), dim3(256), 0, stream>>>(
      w_qkv, w_out, wq_bf, wo_bf);
  win_attn_kernel<<<dim3(4096), dim3(256), 0, stream>>>(
      x, wq_bf, wo_bf, b_out, out);
}
